// LstmCellWithProjection_10565619549070
// MI455X (gfx1250) — compile-verified, matrix operand order checked by reading, not run on hardware
//
#include <hip/hip_runtime.h>
#include <hip/hip_bf16.h>
#include <math.h>

// ---------------------------------------------------------------------------
// LstmCellWithProjection (ELMo cell), B=64 T=256 I=512 H=512 C=4096, clip=3.
//
// MI455X reasoning:
//  * Weights dominate bytes: Wi+Ws+Wp = 72 MB fp32 -> 36 MB bf16, fully
//    L2-resident (192 MB). The 256 sequential steps never touch HBM for
//    weights -> the recurrence is compute/latency bound, so use the fattest
//    matrix op: v_wmma_f32_16x16x32_bf16 (8x the MACs/instr of the f32 WMMA),
//    f32 accumulation.
//  * Per-step math (~2.4 GFLOP) is < 1 us at the bf16 WMMA ceiling, so
//    per-step kernel launches (~1-3 us each, 512 of them) would dominate.
//    => ONE persistent kernel loops over all T steps, with a grid-wide
//    sense-reversal barrier (agent-scope atomics + s_sleep spin) between the
//    gates phase and the projection phase. Grid = multiProcessorCount blocks
//    of 256 threads (8 waves, 8 KB LDS): one block per CU always fits, so all
//    blocks are co-resident and the barrier cannot deadlock.
//  * Gates GEMM fuses [x_t | h] @ [Wi;Ws]^T (K=1024; the 512 boundary is
//    32-aligned so every k-slab reads one source). 4 waves/unit (one gate
//    each) + LDS cross-gate exchange -> fused sigmoid/tanh/clip/masked cell
//    update. Projection (K=4096) is 4-way K-split + LDS reduce, fused
//    clip/mask/h-update/y-write.  Workspace ~56 MB.
// ---------------------------------------------------------------------------

typedef __attribute__((ext_vector_type(16))) __bf16 v16bf;
typedef __attribute__((ext_vector_type(8)))  float  v8f;
typedef __attribute__((ext_vector_type(4)))  unsigned int u32x4;

union TileAB { v16bf v; u32x4 u[2]; };

static constexpr int   Bsz   = 64;
static constexpr int   Tlen  = 256;
static constexpr int   Isz   = 512;
static constexpr int   Hsz   = 512;
static constexpr int   Csz   = 4096;
static constexpr int   Kcomb = Isz + Hsz;   // 1024
static constexpr float CLIPV = 3.0f;

// ----------------------------- device helpers ------------------------------

__device__ __forceinline__ float fexp2(float x) {
#if __has_builtin(__builtin_amdgcn_exp2f)
  return __builtin_amdgcn_exp2f(x);
#else
  return exp2f(x);
#endif
}
__device__ __forceinline__ float frcp(float x) {
#if __has_builtin(__builtin_amdgcn_rcpf)
  return __builtin_amdgcn_rcpf(x);
#else
  return 1.0f / x;
#endif
}
__device__ __forceinline__ float fsigmoid(float x) {
  return frcp(1.0f + fexp2(-1.4426950408889634f * x));
}
__device__ __forceinline__ float ftanh(float x) {
  return 2.0f * frcp(1.0f + fexp2(-2.8853900817779268f * x)) - 1.0f;
}
__device__ __forceinline__ float clip3(float x) {
  return fminf(fmaxf(x, -CLIPV), CLIPV);
}
__device__ __forceinline__ unsigned short f2bf(float f) {
  unsigned int u = __float_as_uint(f);
  unsigned int r = u + 0x7FFFu + ((u >> 16) & 1u);   // round-to-nearest-even
  return (unsigned short)(r >> 16);
}
__device__ __forceinline__ v8f zero8() {
  v8f v;
  #pragma unroll
  for (int i = 0; i < 8; ++i) v[i] = 0.0f;
  return v;
}
__device__ __forceinline__ u32x4 ld8bf(const unsigned short* p) {
  return *(const u32x4*)p;    // 16B load of 8 contiguous bf16
}

// Grid-wide sense-reversal barrier. Safe because grid <= #CUs and each block
// fits one CU => all blocks co-resident. Fences give agent-scope
// release/acquire around the L2 atomic; waiters nap on s_sleep.
__device__ __forceinline__ void grid_barrier(unsigned* cnt, unsigned* sense,
                                             unsigned nblk, unsigned& phase) {
  phase += 1;
  __threadfence();            // release my stores device-wide
  __syncthreads();            // all waves of block done + fenced
  if (threadIdx.x == 0) {
    unsigned prev = __hip_atomic_fetch_add(cnt, 1u, __ATOMIC_ACQ_REL,
                                           __HIP_MEMORY_SCOPE_AGENT);
    if (prev + 1 == nblk) {
      __hip_atomic_store(cnt, 0u, __ATOMIC_RELAXED, __HIP_MEMORY_SCOPE_AGENT);
      __hip_atomic_store(sense, phase, __ATOMIC_RELEASE,
                         __HIP_MEMORY_SCOPE_AGENT);
    } else {
      while (__hip_atomic_load(sense, __ATOMIC_ACQUIRE,
                               __HIP_MEMORY_SCOPE_AGENT) < phase) {
        __builtin_amdgcn_s_sleep(2);
      }
    }
  }
  __syncthreads();
  __threadfence();            // acquire: invalidate near caches before reads
}

// ----------------------------- prologue kernels ----------------------------

__global__ void k_f32_to_bf16(const float* __restrict__ in,
                              unsigned short* __restrict__ out, int n) {
  int i = blockIdx.x * blockDim.x + threadIdx.x;
  if (i < n) out[i] = f2bf(in[i]);
}

// Wcomb[n][k] : k<512 -> w_input[n][k], else w_state[n][k-512]; bf16 row-major
__global__ void k_build_wcomb(const float* __restrict__ wi,
                              const float* __restrict__ wst,
                              unsigned short* __restrict__ out) {
  int i = blockIdx.x * blockDim.x + threadIdx.x;
  if (i >= 4 * Csz * Kcomb) return;
  int n = i >> 10;
  int k = i & (Kcomb - 1);
  float v = (k < Isz) ? wi[(size_t)n * Isz + k]
                      : wst[(size_t)n * Hsz + (k - Isz)];
  out[i] = f2bf(v);
}

__global__ void k_init_state(float* __restrict__ c, float* __restrict__ hf,
                             unsigned short* __restrict__ hb,
                             unsigned* __restrict__ bar) {
  int i = blockIdx.x * blockDim.x + threadIdx.x;
  if (i < Bsz * Csz) c[i] = 0.0f;
  if (i < Bsz * Hsz) { hf[i] = 0.0f; hb[i] = 0; }
  if (i < 64) bar[i] = 0u;    // barrier counter + sense
}

// ----------------------------- persistent kernel ---------------------------
//
// WMMA tile conventions (CDNA5 ISA 7.12.2, wave32):
//   A 16x32 bf16 : lane = khalf*16 + m ; elems j<8 -> k=khalf*8+j,
//                  j>=8 -> k=16+khalf*8+(j-8)  (two 16B loads per lane)
//   B 32x16 bf16 : same k mapping with lane%16 = output column n, and
//                  B[k][n] = W[n][k] (weights row-major over k).
//   C/D f32      : elem r, lanes0-15 -> M=r, lanes16-31 -> M=8+r, N=lane%16.
//
// Block = 256 threads = 8 waves = 2 tile-slots of 4 waves.

__global__ void __launch_bounds__(256)
k_lstm_persistent(const unsigned short* __restrict__ Xb,     // [B,T,I] bf16
                  const unsigned short* __restrict__ Wcomb,  // [4C,1024] bf16
                  const unsigned short* __restrict__ Wpb,    // [H,C] bf16
                  const float* __restrict__ bias,            // [4C]
                  const int* __restrict__ lens,              // [B]
                  float* __restrict__ c,                     // [B,C] f32
                  float* __restrict__ hf,                    // [B,H] f32
                  unsigned short* __restrict__ hb,           // [B,H] bf16
                  unsigned short* __restrict__ act,          // [B,C] bf16
                  float* __restrict__ out,                   // [B,T,H]+[B,H]+[B,C]
                  unsigned* __restrict__ bar) {
  const int tid   = threadIdx.x;
  const int lane  = tid & 31;
  const int wave  = tid >> 5;
  const int slot  = wave >> 2;          // 0..1
  const int gid   = wave & 3;           // gate (phase A) / K-chunk (phase B)
  const int rc    = lane & 15;
  const int khalf = lane >> 4;
  const int tids  = tid & 127;          // thread id within slot
  const unsigned nblk    = gridDim.x;
  const unsigned ustride = nblk * 2;
  const unsigned ubase   = blockIdx.x * 2 + slot;
  unsigned* bar_cnt   = bar;
  unsigned* bar_sense = bar + 16;       // separate cachelines-ish
  unsigned phase = 0;

  __shared__ float red[2][4][8][32];    // [slot][gate/chunk][r][lane]

  for (int t = 0; t < Tlen; ++t) {
    // ---------------- phase A: gates + cell update -------------------------
    // units: 1024 = (Csz/16) x (Bsz/16); 4 waves per unit (one per gate)
    for (unsigned u = ubase; u < (unsigned)(Csz / 16) * (Bsz / 16);
         u += ustride) {
      const int cc = u & 255;           // C-tile
      const int mt = u >> 8;            // M-tile
      const int brow = mt * 16 + rc;
      const unsigned short* xrow = Xb + ((size_t)brow * Tlen + t) * Isz;
      const unsigned short* hrow = hb + (size_t)brow * Hsz;
      const unsigned short* wrow =
          Wcomb + ((size_t)(gid * Csz + cc * 16 + rc)) * Kcomb;

      v8f acc = zero8();
      #pragma unroll 4
      for (int kk = 0; kk < Kcomb / 32; ++kk) {
        const int k0 = kk * 32;
        // k-slabs never straddle the 512 x/h boundary (k0 is 32-aligned)
        const unsigned short* ap =
            ((k0 < Isz) ? (xrow + k0) : (hrow + (k0 - Isz))) + khalf * 8;
        TileAB a, b;
        a.u[0] = ld8bf(ap);
        a.u[1] = ld8bf(ap + 16);
        const unsigned short* bp = wrow + k0 + khalf * 8;
        b.u[0] = ld8bf(bp);
        b.u[1] = ld8bf(bp + 16);
        __builtin_prefetch(bp + 64, 0, 1);   // global_prefetch_b8
        acc = __builtin_amdgcn_wmma_f32_16x16x32_bf16(
            false, a.v, false, b.v, (short)0, acc, false, false);
      }

      #pragma unroll
      for (int r = 0; r < 8; ++r) red[slot][gid][r][lane] = acc[r];
      __syncthreads();

      // fused LSTM cell epilogue: 256 elems / 128 threads of this slot
      for (int e = tids; e < 256; e += 128) {
        const int row = e >> 4, col = e & 15;
        const int li = (row >= 8 ? 16 : 0) + col;
        const int ri = row & 7;
        const int b = mt * 16 + row;
        const int n = cc * 16 + col;
        const size_t idx = (size_t)b * Csz + n;
        float ig = fsigmoid(red[slot][0][ri][li] + bias[0 * Csz + n]);
        float fg = fsigmoid(red[slot][1][ri][li] + bias[1 * Csz + n]);
        float zg = ftanh   (red[slot][2][ri][li] + bias[2 * Csz + n]);
        float og = fsigmoid(red[slot][3][ri][li] + bias[3 * Csz + n]);
        float cold = c[idx];
        float mem = clip3(ig * zg + fg * cold);
        bool active = t < lens[b];
        c[idx] = active ? mem : cold;          // masked cell update
        act[idx] = f2bf(og * ftanh(mem));      // projection input (bf16)
      }
      __syncthreads();                         // red reuse next iteration
    }

    grid_barrier(bar_cnt, bar_sense, nblk, phase);   // act complete

    // ---------------- phase B: projection ----------------------------------
    // units: 128 = (Hsz/16) x (Bsz/16); 4 waves per unit split K=4096
    for (unsigned u = ubase; u < (unsigned)(Hsz / 16) * (Bsz / 16);
         u += ustride) {
      const int nt = u & 31;            // H-tile
      const int mt = u >> 5;            // M-tile
      const unsigned short* arow = act + (size_t)(mt * 16 + rc) * Csz;
      const unsigned short* wrow = Wpb + (size_t)(nt * 16 + rc) * Csz;
      const int kbeg = gid * (Csz / 4);

      v8f acc = zero8();
      #pragma unroll 4
      for (int kk = 0; kk < (Csz / 4) / 32; ++kk) {
        const int k0 = kbeg + kk * 32 + khalf * 8;
        TileAB a, b;
        a.u[0] = ld8bf(arow + k0);
        a.u[1] = ld8bf(arow + k0 + 16);
        b.u[0] = ld8bf(wrow + k0);
        b.u[1] = ld8bf(wrow + k0 + 16);
        __builtin_prefetch(wrow + k0 + 64, 0, 1);
        acc = __builtin_amdgcn_wmma_f32_16x16x32_bf16(
            false, a.v, false, b.v, (short)0, acc, false, false);
      }

      #pragma unroll
      for (int r = 0; r < 8; ++r) red[slot][gid][r][lane] = acc[r];
      __syncthreads();

      for (int e = tids; e < 256; e += 128) {
        const int row = e >> 4, col = e & 15;
        const int li = (row >= 8 ? 16 : 0) + col;
        const int ri = row & 7;
        float s = red[slot][0][ri][li] + red[slot][1][ri][li] +
                  red[slot][2][ri][li] + red[slot][3][ri][li];
        float outv = clip3(s);
        const int b = mt * 16 + row;
        const int n = nt * 16 + col;
        bool active = t < lens[b];
        const size_t hidx = (size_t)b * Hsz + n;
        if (active) {                          // masked h update (else frozen)
          hf[hidx] = outv;
          hb[hidx] = f2bf(outv);
        }
        out[((size_t)b * Tlen + t) * Hsz + n] = active ? outv : 0.0f;
      }
      __syncthreads();
    }

    grid_barrier(bar_cnt, bar_sense, nblk, phase);   // h complete
  }

  // ---------------- finalize: (h_f, c_f) into output tail -------------------
  const unsigned gtid = blockIdx.x * blockDim.x + tid;
  const unsigned gtot = nblk * blockDim.x;
  const size_t yoff = (size_t)Bsz * Tlen * Hsz;
  for (unsigned i = gtid; i < Bsz * Hsz; i += gtot) out[yoff + i] = hf[i];
  for (unsigned i = gtid; i < Bsz * Csz; i += gtot)
    out[yoff + Bsz * Hsz + i] = c[i];
}

// ----------------------------- host ----------------------------------------

extern "C" void kernel_launch(void* const* d_in, const int* in_sizes, int n_in,
                              void* d_out, int out_size, void* d_ws,
                              size_t ws_size, hipStream_t stream) {
  (void)in_sizes; (void)n_in; (void)out_size; (void)ws_size;
  const float* inputs  = (const float*)d_in[0];
  const int*   lens    = (const int*)  d_in[1];
  const float* w_input = (const float*)d_in[2];
  const float* w_state = (const float*)d_in[3];
  const float* bias    = (const float*)d_in[4];
  const float* w_proj  = (const float*)d_in[5];
  float* out = (float*)d_out;

  // workspace layout (256B aligned chunks), total ~56 MB
  char* ws = (char*)d_ws;
  size_t off = 0;
  auto take = [&](size_t bytes) {
    void* p = ws + off;
    off += (bytes + 255) & ~(size_t)255;
    return p;
  };
  unsigned short* Wcomb = (unsigned short*)take((size_t)4 * Csz * Kcomb * 2); // 32 MB
  unsigned short* Wpb   = (unsigned short*)take((size_t)Hsz * Csz * 2);       //  4 MB
  unsigned short* Xb    = (unsigned short*)take((size_t)Bsz * Tlen * Isz * 2);// 16 MB
  float*          cst   = (float*)         take((size_t)Bsz * Csz * 4);       //  1 MB
  float*          hf    = (float*)         take((size_t)Bsz * Hsz * 4);
  unsigned short* hb    = (unsigned short*)take((size_t)Bsz * Hsz * 2);
  unsigned short* actb  = (unsigned short*)take((size_t)Bsz * Csz * 2);
  unsigned*       bar   = (unsigned*)      take(256);

  // persistent grid: one block per CU guarantees co-residency (no deadlock)
  int dev = 0;
  (void)hipGetDevice(&dev);
  int nblk = 0;
  if (hipDeviceGetAttribute(&nblk, hipDeviceAttributeMultiprocessorCount,
                            dev) != hipSuccess || nblk <= 0)
    nblk = 64;
  if (nblk > 1024) nblk = 1024;

  // prologue: bf16 conversions + zero state/barrier (re-done every call)
  {
    int n = Bsz * Tlen * Isz;
    k_f32_to_bf16<<<(n + 255) / 256, 256, 0, stream>>>(inputs, Xb, n);
  }
  {
    int n = 4 * Csz * Kcomb;
    k_build_wcomb<<<(n + 255) / 256, 256, 0, stream>>>(w_input, w_state, Wcomb);
  }
  {
    int n = Hsz * Csz;
    k_f32_to_bf16<<<(n + 255) / 256, 256, 0, stream>>>(w_proj, Wpb, n);
  }
  {
    int n = Bsz * Csz;
    k_init_state<<<(n + 255) / 256, 256, 0, stream>>>(cst, hf, hb, bar);
  }

  // whole recurrence + finalize in ONE persistent kernel
  k_lstm_persistent<<<nblk, 256, 0, stream>>>(Xb, Wcomb, Wpb, bias, lens, cst,
                                              hf, hb, actb, out, bar);
}
